// XMultiHeadAttention_84550726189715
// MI455X (gfx1250) — matrix operand order repaired and checked by reading, not compiled, not run on hardware
//
#include <hip/hip_runtime.h>

// ---------------------------------------------------------------------------
// Graphormer-style attention, MI455X (gfx1250, wave32, WMMA 16x16x32 f16,
// async global->LDS staging, ASYNCcnt software pipelining)
// ---------------------------------------------------------------------------
#define B_  16
#define N_  512
#define HID_ 1024
#define H_  16
#define D_  64
#define TH_ 32
#define TE_ 64
#define SCALE_ 0.125f

typedef __attribute__((ext_vector_type(16))) _Float16 v16h;
typedef __attribute__((ext_vector_type(8)))  _Float16 v8h;
typedef __attribute__((ext_vector_type(8)))  float    v8f;
typedef int v4i_t __attribute__((vector_size(16)));

#define DEV static __device__ __forceinline__

#if defined(__gfx1250__) && __has_builtin(__builtin_amdgcn_global_load_async_to_lds_b128)
#define USE_ASYNC_LDS 1
#else
#define USE_ASYNC_LDS 0
#endif

// 16-byte global -> LDS copy: async (ASYNCcnt) when available, sync fallback.
DEV void cp16(void* lds, const void* g) {
#if USE_ASYNC_LDS
  __builtin_amdgcn_global_load_async_to_lds_b128(
      (__attribute__((address_space(1))) v4i_t*)g,
      (__attribute__((address_space(3))) v4i_t*)lds, 0, 0);
#else
  *(v8h*)lds = *(const v8h*)g;
#endif
}

template <int CNT> DEV void wait_async() {
#if USE_ASYNC_LDS
#if __has_builtin(__builtin_amdgcn_s_wait_asynccnt)
  __builtin_amdgcn_s_wait_asynccnt(CNT);
#else
  asm volatile("s_wait_asynccnt %0" ::"i"(CNT) : "memory");
#endif
#endif
}

DEV v8f zf8() { v8f z = {0.f,0.f,0.f,0.f,0.f,0.f,0.f,0.f}; return z; }

DEV v8f wmma_f16(v16h a, v16h b, v8f c) {
  return __builtin_amdgcn_wmma_f32_16x16x32_f16(false, a, false, b, (short)0, c,
                                                false, false);
}

// A-fragment 16x32 f16 from row-major LDS tile (ldm halves/row).
DEV v16h ldsA(const _Float16* p0, int row0, int k0, int ldm, int lane) {
  const int m = lane & 15, hi = lane >> 4;
  const _Float16* p = p0 + (row0 + m) * ldm + k0 + hi * 8;
  v8h lo = *(const v8h*)p;          // k = hi*8 + 0..7
  v8h hv = *(const v8h*)(p + 16);   // k = 16 + hi*8 + 0..7
  return __builtin_shufflevector(lo, hv, 0, 1, 2, 3, 4, 5, 6, 7,
                                 8, 9, 10, 11, 12, 13, 14, 15);
}

// B-fragment 32x16 where B[k,n] = src[n][k] (each lane reads a contiguous row).
DEV v16h ldsB_rowk(const _Float16* p0, int n0, int k0, int ldm, int lane) {
  const int n = lane & 15, hi = lane >> 4;
  const _Float16* p = p0 + (n0 + n) * ldm + k0 + hi * 8;
  v8h lo = *(const v8h*)p;
  v8h hv = *(const v8h*)(p + 16);
  return __builtin_shufflevector(lo, hv, 0, 1, 2, 3, 4, 5, 6, 7,
                                 8, 9, 10, 11, 12, 13, 14, 15);
}

// ---------------------------------------------------------------------------
// f32 -> f16 bulk conversion (8 elems/thread)
// ---------------------------------------------------------------------------
__global__ __launch_bounds__(256) void cvt16_kernel(
    const float* __restrict__ src, _Float16* __restrict__ dst) {
  size_t i = ((size_t)blockIdx.x * 256 + threadIdx.x) * 8;
  float4 f0 = *(const float4*)(src + i);
  float4 f1 = *(const float4*)(src + i + 4);
  v8h o;
  o[0] = (_Float16)f0.x; o[1] = (_Float16)f0.y;
  o[2] = (_Float16)f0.z; o[3] = (_Float16)f0.w;
  o[4] = (_Float16)f1.x; o[5] = (_Float16)f1.y;
  o[6] = (_Float16)f1.z; o[7] = (_Float16)f1.w;
  *(v8h*)(dst + i) = o;
}

// ---------------------------------------------------------------------------
// int32 -> u8 index compaction (4x less bias-gather traffic)
// ---------------------------------------------------------------------------
__global__ __launch_bounds__(256) void idx8_kernel(
    const int* __restrict__ hop, const int* __restrict__ edg,
    unsigned char* __restrict__ h8, unsigned char* __restrict__ e8) {
  size_t i = (size_t)blockIdx.x * 256 + threadIdx.x;
  h8[i] = (unsigned char)hop[i];
  e8[i] = (unsigned char)edg[i];
}

// ---------------------------------------------------------------------------
// GEMM: C[m,col] = A_f16[m,:] . W_f16[col,:] + bias[col]
// M=8192, N=1024, K=1024. Tile 128x64, K-step 64, async double-buffered LDS.
// MODE 0: f16 out scattered to [B,H,N,D]   (q, k)
// MODE 1: f16 out scattered to [B,H,D,N]   (v, transposed for P@V B-frags)
// MODE 2: f32 out row-major [M,1024]       (final projection)
// ---------------------------------------------------------------------------
template <int MODE>
__global__ __launch_bounds__(256) void gemm16_kernel(
    const _Float16* __restrict__ A, const _Float16* __restrict__ W,
    const float* __restrict__ bias, void* __restrict__ outp) {
  const int M0 = blockIdx.x * 128, N0 = blockIdx.y * 64;
  const int tid = threadIdx.x, lane = tid & 31, wave = tid >> 5;
  const int mw = wave >> 1, nw = wave & 1;   // 4x2 wave grid, 32x32 per wave

  __shared__ __align__(16) _Float16 aS[2][128 * 64];  // 32 KB
  __shared__ __align__(16) _Float16 wS[2][64 * 64];   // 16 KB

  auto stage = [&](int buf, int k0) {
#pragma unroll
    for (int t = 0; t < 4; ++t) {              // 128x64 halves = 1024 x b128
      int s = tid + t * 256;
      int r = s >> 3, c = (s & 7) * 8;
      cp16(&aS[buf][r * 64 + c], A + (size_t)(M0 + r) * HID_ + k0 + c);
    }
#pragma unroll
    for (int t = 0; t < 2; ++t) {              // 64x64 halves = 512 x b128
      int s = tid + t * 256;
      int r = s >> 3, c = (s & 7) * 8;
      cp16(&wS[buf][r * 64 + c], W + (size_t)(N0 + r) * HID_ + k0 + c);
    }
  };

  v8f acc[2][2];
#pragma unroll
  for (int i = 0; i < 2; ++i)
#pragma unroll
    for (int j = 0; j < 2; ++j) acc[i][j] = zf8();

  stage(0, 0);
  wait_async<0>();
  __syncthreads();
  int cur = 0;
  for (int k0 = 0; k0 < HID_; k0 += 64) {
    if (k0 + 64 < HID_) stage(cur ^ 1, k0 + 64);   // overlap next stage
#pragma unroll
    for (int kk = 0; kk < 2; ++kk)
#pragma unroll
      for (int im = 0; im < 2; ++im) {
        v16h a = ldsA(aS[cur], mw * 32 + im * 16, kk * 32, 64, lane);
#pragma unroll
        for (int in = 0; in < 2; ++in) {
          v16h bf = ldsB_rowk(wS[cur], nw * 32 + in * 16, kk * 32, 64, lane);
          acc[im][in] = wmma_f16(a, bf, acc[im][in]);
        }
      }
    wait_async<0>();
    __syncthreads();
    cur ^= 1;
  }

  const int cn = lane & 15, hi = lane >> 4;
#pragma unroll
  for (int im = 0; im < 2; ++im)
#pragma unroll
    for (int in = 0; in < 2; ++in) {
      int col = N0 + nw * 32 + in * 16 + cn;
      float bv = bias[col];
#pragma unroll
      for (int v = 0; v < 8; ++v) {
        int m = M0 + mw * 32 + im * 16 + v + 8 * hi;
        if constexpr (MODE == 0) {
          int b = m >> 9, i = m & 511, h = col >> 6, d = col & 63;
          ((_Float16*)outp)[(((size_t)b * H_ + h) * N_ + i) * D_ + d] =
              (_Float16)(acc[im][in][v] + bv);
        } else if constexpr (MODE == 1) {
          int b = m >> 9, i = m & 511, h = col >> 6, d = col & 63;
          ((_Float16*)outp)[(((size_t)b * H_ + h) * D_ + d) * N_ + i] =
              (_Float16)(acc[im][in][v] + bv);
        } else {
          ((float*)outp)[(size_t)m * HID_ + col] = acc[im][in][v] + bv;
        }
      }
    }
}

// ---------------------------------------------------------------------------
// Score tables: s_hop[row,t] = q[row]·qhe[t,h] + k[row]·khe[t,h]   (t<32)
//               s_edge[row,t]= q[row]·qee[t,h] + k[row]·kee[t,h]   (t<64)
// row = (b*H + h)*N + i.  One thread per (row, t-combined<96).
// ---------------------------------------------------------------------------
__global__ __launch_bounds__(256) void score_kernel(
    const _Float16* __restrict__ q, const _Float16* __restrict__ k,
    const float* __restrict__ qhe, const float* __restrict__ khe,
    const float* __restrict__ qee, const float* __restrict__ kee,
    float* __restrict__ s_hop, float* __restrict__ s_edge) {
  size_t idx = (size_t)blockIdx.x * 256 + threadIdx.x;
  int tc = (int)(idx % 96);
  size_t row = idx / 96;
  int h = (int)((row >> 9) & (H_ - 1));
  const _Float16* qp = q + row * D_;
  const _Float16* kp = k + row * D_;
  if (tc < TH_) {
    const float* e1 = qhe + (size_t)tc * HID_ + h * D_;
    const float* e2 = khe + (size_t)tc * HID_ + h * D_;
    float s = 0.f;
#pragma unroll 8
    for (int d = 0; d < D_; ++d)
      s += (float)qp[d] * e1[d] + (float)kp[d] * e2[d];
    s_hop[row * TH_ + tc] = s;
  } else {
    int t = tc - TH_;
    const float* e1 = qee + (size_t)t * HID_ + h * D_;
    const float* e2 = kee + (size_t)t * HID_ + h * D_;
    float s = 0.f;
#pragma unroll 8
    for (int d = 0; d < D_; ++d)
      s += (float)qp[d] * e1[d] + (float)kp[d] * e2[d];
    s_edge[row * TE_ + t] = s;
  }
}

// ---------------------------------------------------------------------------
// Fused attention (flash-style online softmax) + bucket scatter + x2/x3.
// Grid: B*H*(N/32); 256 threads = 8 waves; wave w -> (mt=w>>2, dt=w&3) owns
// the 16x16 tile (rows mt*16, cols dt*16) of both the S-chunk and x1.
// ASYNCcnt software pipeline (loads complete in order):
//   prologue: issue q/tables(4), issue k(0)(2); wait<=2 -> tables ready
//   iter jc : issue v(jc)(2); wait<=2 -> k(jc) ready; S-WMMA;
//             issue k(jc+1)(2); softmax/scatter phase;
//             wait<=2 (or 0 on last) -> v(jc) ready; P@V-WMMA.
// v is stored transposed [B,H,D,N] so P@V B-frags are contiguous rowk loads.
// ---------------------------------------------------------------------------
__global__ __launch_bounds__(256) void attn_kernel(
    const _Float16* __restrict__ qf, const _Float16* __restrict__ kf,
    const _Float16* __restrict__ vt,
    const float* __restrict__ s_hop, const float* __restrict__ s_edge,
    const unsigned char* __restrict__ hop8, const unsigned char* __restrict__ edg8,
    const float* __restrict__ vhe_emb, const float* __restrict__ vee_emb,
    _Float16* __restrict__ y) {
  const int blk = blockIdx.x;
  const int ib = blk & 15, h = (blk >> 4) & 15, b = blk >> 8;
  const int i0 = ib * 32;
  const int tid = threadIdx.x, lane = tid & 31, wave = tid >> 5;
  const int mt = wave >> 2, dt = wave & 3;
  const size_t bh = (size_t)b * H_ + h;
  const size_t rowbase = bh * N_ + i0;

  __shared__ __align__(16) _Float16 qA[32 * 64];     // 4 KB
  __shared__ __align__(16) float shop[32 * TH_];     // 4 KB
  __shared__ __align__(16) float sedg[32 * TE_];     // 8 KB
  __shared__ __align__(16) float bha[32 * TH_];      // 4 KB hop buckets
  __shared__ __align__(16) float bea[32 * TE_];      // 8 KB edge buckets
  __shared__ float mrow[32], lrow[32], rrow[32];
  __shared__ float red[256];                         // 1 KB
  __shared__ __align__(16) unsigned char ubuf[32768];// reused region
  _Float16* kv  = (_Float16*)ubuf;                   // 64x64 f16 k chunk
  float*    Sch = (float*)(ubuf + 8192);             // 32x64 f32 logits chunk
  _Float16* Pch = (_Float16*)(ubuf + 16384);         // 32x64 f16 probs chunk
  _Float16* vSt = (_Float16*)(ubuf + 20480);         // 64(d)x64(j) f16 v^T chunk

  { // q tile (async)
    int r = tid >> 3, c = (tid & 7) * 8;
    cp16(qA + r * 64 + c, qf + (rowbase + r) * D_ + c);
  }
  { // per-row hop score table
    int r = tid >> 3, c = (tid & 7) * 4;
    cp16(shop + r * TH_ + c, s_hop + (rowbase + r) * TH_ + c);
  }
#pragma unroll
  for (int t = 0; t < 2; ++t) { // per-row edge score table
    int s = tid + t * 256;
    int r = s >> 4, c = (s & 15) * 4;
    cp16(sedg + r * TE_ + c, s_edge + (rowbase + r) * TE_ + c);
  }
#pragma unroll
  for (int t = 0; t < 2; ++t) { // prefetch k(0)
    int s = tid + t * 256;
    int r = s >> 3, c = (s & 7) * 8;
    cp16(kv + r * 64 + c, kf + (bh * N_ + r) * D_ + c);
  }
  for (int s = tid; s < 32 * 96; s += 256) { // zero buckets
    if (s < 1024) bha[s] = 0.f; else bea[s - 1024] = 0.f;
  }
  if (tid < 32) { mrow[tid] = -1e30f; lrow[tid] = 0.f; }

  v8f accX = zf8();
  wait_async<2>();             // q/shop/sedg landed; k(0) still in flight
  __syncthreads();

  const size_t idxbase = ((size_t)b * N_ + i0) * N_;

  for (int jc = 0; jc < N_; jc += 64) {
    // ---- issue v^T(jc) chunk: rows are d (64), cols are j (64) ----
#pragma unroll
    for (int t = 0; t < 2; ++t) {
      int s = tid + t * 256;
      int r = s >> 3, c = (s & 7) * 8;
      cp16(vSt + r * 64 + c, vt + (bh * D_ + r) * N_ + jc + c);
    }
    wait_async<2>();           // k(jc) done (in-order); v(jc) in flight
    __syncthreads();

    // ---- S tile: q[mt*16..] x k[dt*16..]^T (K = D = 64) ----
    {
      v8f s = zf8();
#pragma unroll
      for (int kk = 0; kk < 2; ++kk) {
        v16h a  = ldsA(qA, mt * 16, kk * 32, 64, lane);
        v16h bf = ldsB_rowk(kv, dt * 16, kk * 32, 64, lane);
        s = wmma_f16(a, bf, s);
      }
      int n = lane & 15, hi = lane >> 4;
#pragma unroll
      for (int v = 0; v < 8; ++v)
        Sch[(mt * 16 + v + 8 * hi) * 64 + dt * 16 + n] = s[v];
    }
    __syncthreads();

    // ---- prefetch k(jc+1) into kv (its last reader was the S step) ----
    if (jc + 64 < N_) {
#pragma unroll
      for (int t = 0; t < 2; ++t) {
        int s = tid + t * 256;
        int r = s >> 3, c = (s & 7) * 8;
        cp16(kv + r * 64 + c, kf + (bh * N_ + jc + 64 + r) * D_ + c);
      }
    }

    // ---- bias gather + scale + chunk max ----
    {
      int row = tid >> 3, c0 = (tid & 7) * 8;
      const unsigned char* hp = hop8 + idxbase + (size_t)row * N_ + jc + c0;
      const unsigned char* ep = edg8 + idxbase + (size_t)row * N_ + jc + c0;
      float lmax = -1e30f;
#pragma unroll
      for (int u = 0; u < 8; ++u) {
        float val = (Sch[row * 64 + c0 + u] + shop[row * TH_ + hp[u]] +
                     sedg[row * TE_ + ep[u]]) * SCALE_;
        Sch[row * 64 + c0 + u] = val;
        lmax = fmaxf(lmax, val);
      }
      red[tid] = lmax;
    }
    __syncthreads();
    if (tid < 32) {
      float cm = red[tid * 8];
#pragma unroll
      for (int u = 1; u < 8; ++u) cm = fmaxf(cm, red[tid * 8 + u]);
      float mnew = fmaxf(mrow[tid], cm);
      float r = __expf(mrow[tid] - mnew);
      rrow[tid] = r; lrow[tid] *= r; mrow[tid] = mnew;
    }
    __syncthreads();

    // ---- rescale buckets by exp(m_old - m_new) ----
    for (int s = tid; s < 32 * 96; s += 256) {
      if (s < 1024) bha[s] *= rrow[s >> 5];
      else { int t = s - 1024; bea[t] *= rrow[t >> 6]; }
    }
    __syncthreads();

    // ---- exp, row partial sums, P(f16), bucket scatter (ds_add_f32) ----
    {
      int row = tid >> 3, c0 = (tid & 7) * 8;
      const unsigned char* hp = hop8 + idxbase + (size_t)row * N_ + jc + c0;
      const unsigned char* ep = edg8 + idxbase + (size_t)row * N_ + jc + c0;
      float mn = mrow[row], ps = 0.f;
#pragma unroll
      for (int u = 0; u < 8; ++u) {
        float p = __expf(Sch[row * 64 + c0 + u] - mn);
        ps += p;
        Pch[row * 64 + c0 + u] = (_Float16)p;
        atomicAdd(&bha[row * TH_ + hp[u]], p);
        atomicAdd(&bea[row * TE_ + ep[u]], p);
      }
      red[tid] = ps;
    }
    __syncthreads();
    if (tid < 32) {
      float s = 0.f;
#pragma unroll
      for (int u = 0; u < 8; ++u) s += red[tid * 8 + u];
      lrow[tid] += s;
    }
    { // rescale x1 accumulators (C layout: row = v + 8*hi)
      int hi = lane >> 4;
#pragma unroll
      for (int v = 0; v < 8; ++v) accX[v] *= rrow[mt * 16 + v + 8 * hi];
    }
    if (jc + 64 < N_) wait_async<2>();  // v(jc) done; k(jc+1) in flight
    else              wait_async<0>();  // last: only v(jc) outstanding
    __syncthreads();

    // ---- accX += P_chunk @ V_chunk (v^T rows are d -> contiguous B) ----
#pragma unroll
    for (int kk = 0; kk < 2; ++kk) {
      v16h a  = ldsA(Pch, mt * 16, kk * 32, 64, lane);
      v16h bf = ldsB_rowk(vSt, dt * 16, kk * 32, 64, lane);
      accX = wmma_f16(a, bf, accX);
    }
    __syncthreads();
  }

  // ---- epilogue: normalize, bucket GEMMs (x2+x3), write y f16 ----
  if (tid < 32) lrow[tid] = 1.0f / lrow[tid];
  float* vheS = (float*)ubuf;                 // 32x64
  float* veeS = (float*)(ubuf + 8192);        // 64x64
  float* ybuf = (float*)(ubuf + 8192 + 16384);// 32x64
#pragma unroll
  for (int t = 0; t < 2; ++t) {               // vhe[t, h*64 + d] (async)
    int s = tid + t * 256;
    int r = s >> 4, c = (s & 15) * 4;
    cp16(vheS + r * 64 + c, vhe_emb + (size_t)r * HID_ + h * D_ + c);
  }
#pragma unroll
  for (int t = 0; t < 4; ++t) {               // vee[t, h*64 + d] (async)
    int s = tid + t * 256;
    int r = s >> 4, c = (s & 15) * 4;
    cp16(veeS + r * 64 + c, vee_emb + (size_t)r * HID_ + h * D_ + c);
  }
  wait_async<0>();
  __syncthreads();
  for (int s = tid; s < 32 * 96; s += 256) {  // normalize buckets
    if (s < 1024) bha[s] *= lrow[s >> 5];
    else { int t = s - 1024; bea[t] *= lrow[t >> 6]; }
  }
  __syncthreads();
  {
    int row = tid >> 3, d0 = (tid & 7) * 8;
#pragma unroll
    for (int u = 0; u < 8; ++u) {
      int d = d0 + u; float s = 0.f;
      for (int t = 0; t < TH_; ++t) s += bha[row * TH_ + t] * vheS[t * 64 + d];
      for (int t = 0; t < TE_; ++t) s += bea[row * TE_ + t] * veeS[t * 64 + d];
      ybuf[row * 64 + d] = s;
    }
  }
  __syncthreads();
  {
    int n = lane & 15, hi = lane >> 4;
#pragma unroll
    for (int v = 0; v < 8; ++v) {
      int m = mt * 16 + v + 8 * hi;
      int d = dt * 16 + n;
      float val = accX[v] * lrow[m] + ybuf[m * 64 + d];
      y[((size_t)b * N_ + i0 + m) * HID_ + h * D_ + d] = (_Float16)val;
    }
  }
}

// ---------------------------------------------------------------------------
extern "C" void kernel_launch(void* const* d_in, const int* in_sizes, int n_in,
                              void* d_out, int out_size, void* d_ws, size_t ws_size,
                              hipStream_t stream) {
  (void)in_sizes; (void)n_in; (void)out_size; (void)ws_size;
  const float* x   = (const float*)d_in[0];
  const float* qhe = (const float*)d_in[1];
  const float* qee = (const float*)d_in[2];
  const float* khe = (const float*)d_in[3];
  const float* kee = (const float*)d_in[4];
  const float* vhe = (const float*)d_in[5];
  const float* vee = (const float*)d_in[6];
  const int*   hop = (const int*)d_in[7];
  const int*   edg = (const int*)d_in[8];
  const float* Wq  = (const float*)d_in[9];
  const float* bq  = (const float*)d_in[10];
  const float* Wk  = (const float*)d_in[11];
  const float* bk  = (const float*)d_in[12];
  const float* Wv  = (const float*)d_in[13];
  const float* bv  = (const float*)d_in[14];
  const float* Wo  = (const float*)d_in[15];
  const float* bo  = (const float*)d_in[16];

  char* ws = (char*)d_ws;
  size_t off = 0;
  auto take = [&](size_t bytes) {
    char* p = ws + off;
    off += (bytes + 255) & ~(size_t)255;
    return (void*)p;
  };
  const size_t QKV_BYTES = (size_t)B_ * H_ * N_ * D_ * sizeof(_Float16); // 16 MB
  const size_t W_ELEMS   = (size_t)HID_ * HID_;
  _Float16* qf  = (_Float16*)take(QKV_BYTES);
  _Float16* kf  = (_Float16*)take(QKV_BYTES);
  _Float16* vtf = (_Float16*)take(QKV_BYTES);          // v transposed [B,H,D,N]
  _Float16* yf  = (_Float16*)take((size_t)B_ * N_ * HID_ * sizeof(_Float16));
  float*    shp = (float*)take((size_t)B_ * H_ * N_ * TH_ * sizeof(float));
  float*    sed = (float*)take((size_t)B_ * H_ * N_ * TE_ * sizeof(float));
  unsigned char* h8 = (unsigned char*)take((size_t)B_ * N_ * N_);
  unsigned char* e8 = (unsigned char*)take((size_t)B_ * N_ * N_);
  _Float16* xh  = (_Float16*)take((size_t)B_ * N_ * HID_ * sizeof(_Float16));
  _Float16* wqh = (_Float16*)take(W_ELEMS * sizeof(_Float16));
  _Float16* wkh = (_Float16*)take(W_ELEMS * sizeof(_Float16));
  _Float16* wvh = (_Float16*)take(W_ELEMS * sizeof(_Float16));
  _Float16* woh = (_Float16*)take(W_ELEMS * sizeof(_Float16));

  // 0) one-time f32->f16 conversion of x and all weights
  cvt16_kernel<<<((size_t)B_ * N_ * HID_) / 2048, 256, 0, stream>>>(x, xh);
  cvt16_kernel<<<W_ELEMS / 2048, 256, 0, stream>>>(Wq, wqh);
  cvt16_kernel<<<W_ELEMS / 2048, 256, 0, stream>>>(Wk, wkh);
  cvt16_kernel<<<W_ELEMS / 2048, 256, 0, stream>>>(Wv, wvh);
  cvt16_kernel<<<W_ELEMS / 2048, 256, 0, stream>>>(Wo, woh);

  // 1) compact indices to u8
  idx8_kernel<<<(B_ * N_ * N_) / 256, 256, 0, stream>>>(hop, edg, h8, e8);

  // 2) q/k/v projections (WMMA GEMM, f16 out; v written transposed)
  dim3 ggrid(B_ * N_ / 128, HID_ / 64);
  gemm16_kernel<0><<<ggrid, 256, 0, stream>>>(xh, wqh, bq, qf);
  gemm16_kernel<0><<<ggrid, 256, 0, stream>>>(xh, wkh, bk, kf);
  gemm16_kernel<1><<<ggrid, 256, 0, stream>>>(xh, wvh, bv, vtf);

  // 3) per-row hop/edge score tables
  score_kernel<<<((size_t)B_ * H_ * N_ * 96) / 256, 256, 0, stream>>>(
      qf, kf, qhe, khe, qee, kee, shp, sed);

  // 4) fused attention + bucket scatter + x2/x3, writes y f16
  attn_kernel<<<B_ * H_ * (N_ / 32), 256, 0, stream>>>(
      qf, kf, vtf, shp, sed, h8, e8, vhe, vee, yf);

  // 5) output projection -> f32 d_out
  gemm16_kernel<2><<<ggrid, 256, 0, stream>>>(yf, woh, bo, (float*)d_out);
}
